// SelfAttention_19799799234572
// MI455X (gfx1250) — compile-verified, matrix operand order checked by reading, not run on hardware
//
#include <hip/hip_runtime.h>
#include <cstdint>

// ---------------------------------------------------------------------------
// Types for CDNA5 WMMA (wave32): v_wmma_f32_16x16x32_bf16
// ---------------------------------------------------------------------------
typedef __attribute__((ext_vector_type(16))) __bf16 v16bf;
typedef __attribute__((ext_vector_type(8)))  float  v8f;

static __device__ __forceinline__ v8f wmma_bf16(v16bf a, v16bf b, v8f c) {
  return __builtin_amdgcn_wmma_f32_16x16x32_bf16(
      /*neg_a=*/false, a, /*neg_b=*/false, b,
      /*c_mod=*/(short)0, c, /*reuse_a=*/false, /*reuse_b=*/false);
}

// Load a 16-lane-striped bf16 fragment from LDS pointer p (= &tile[row][kb]):
// per ISA 7.12.2, lane holds elems 0..7 -> K=kb..kb+7 and elems 8..15 -> K=kb+16..kb+23.
static __device__ __forceinline__ v16bf load_frag(const __bf16* p) {
  union { v16bf v; uint4 u[2]; } f;
  f.u[0] = *(const uint4*)(p);
  f.u[1] = *(const uint4*)(p + 16);
  return f.v;
}

// Async direct global->LDS copy (16B per lane), tracked by ASYNCcnt.
// LDS byte address = low 32 bits of the generic pointer (aperture spec).
static __device__ __forceinline__ void async_copy_b128(const void* gptr, void* lptr) {
  unsigned lds = (unsigned)(uintptr_t)lptr;
  unsigned long long ga = (unsigned long long)(uintptr_t)gptr;
  asm volatile("global_load_async_to_lds_b128 %0, %1, off"
               :: "v"(lds), "v"(ga) : "memory");
}
#define S_WAIT_ASYNC(n) asm volatile("s_wait_asynccnt " n ::: "memory")

union U8 { uint4 u; __bf16 h[8]; };

// ---------------------------------------------------------------------------
// Kernel 1: fp32 -> bf16 conversion (vectorized x4; all sizes divisible by 4)
// ---------------------------------------------------------------------------
__global__ __launch_bounds__(256) void f32_to_bf16(const float* __restrict__ src,
                                                   __bf16* __restrict__ dst, int n) {
  int i = (blockIdx.x * 256 + threadIdx.x) * 4;
  if (i + 3 < n) {
    float4 f = *(const float4*)(src + i);
    dst[i + 0] = (__bf16)f.x;
    dst[i + 1] = (__bf16)f.y;
    dst[i + 2] = (__bf16)f.z;
    dst[i + 3] = (__bf16)f.w;
  }
}

// ---------------------------------------------------------------------------
// Kernel 2/5: bf16 GEMM  C = A[M,K] * B[K,N] + bias[N]
// Block tile 128x128, BK=32, 256 threads = 8 waves, wave tile 64x32.
// A: async double-buffered direct-to-LDS. B: register-staged, transposed into
// a single LDS buffer in the window between the two barriers.
// ---------------------------------------------------------------------------
#define GBM 128
#define GBN 128
#define GBK 32
#define APAD 40   // 32 + 8 pad (80B row stride -> conflict-free, 16B aligned)

template <bool OUT_F32>
__global__ __launch_bounds__(256) void gemm_bf16(
    const __bf16* __restrict__ A, int lda,
    const __bf16* __restrict__ Bm, int ldb,
    const float* __restrict__ bias,
    __bf16* __restrict__ outb, float* __restrict__ outf, int ldc,
    int K) {
  __shared__ __align__(16) __bf16 As[2][GBM][APAD];
  __shared__ __align__(16) __bf16 Bs[GBN][APAD];  // [n][k] (transposed)

  const int tid  = threadIdx.x;
  const int wave = tid >> 5;
  const int lane = tid & 31;
  const int m0 = blockIdx.y * GBM;
  const int n0 = blockIdx.x * GBN;
  const int wm = (wave >> 2) * 64;  // 2 wave-rows
  const int wn = (wave & 3) * 32;   // 4 wave-cols
  const int kb = (lane < 16) ? 0 : 8;
  const int nlane = lane & 15;
  const int moff = (lane >= 16) ? 8 : 0;

  // A-tile async slots: 128 rows x 32 cols = 512 x 16B chunks, 2 per thread
  const int ar = tid >> 2, ac = (tid & 3) * 8;
  auto issueA = [&](int k0, int bufi) {
    async_copy_b128(&A[(size_t)(m0 + ar) * lda + k0 + ac], &As[bufi][ar][ac]);
    async_copy_b128(&A[(size_t)(m0 + ar + 64) * lda + k0 + ac], &As[bufi][ar + 64][ac]);
  };
  // B-tile register staging: 32 rows x 128 cols, 16 elems per thread
  const int kr = tid >> 3, nc = (tid & 7) * 16;
  U8 rb0, rb1;
  auto loadB = [&](int k0) {
    rb0.u = *(const uint4*)&Bm[(size_t)(k0 + kr) * ldb + n0 + nc];
    rb1.u = *(const uint4*)&Bm[(size_t)(k0 + kr) * ldb + n0 + nc + 8];
  };
  auto storeB = [&]() {
    #pragma unroll
    for (int e = 0; e < 8; ++e) {
      Bs[nc + e][kr]     = rb0.h[e];
      Bs[nc + 8 + e][kr] = rb1.h[e];
    }
  };

  v8f acc[4][2] = {};
  issueA(0, 0);
  loadB(0);

  const int NIT = K / GBK;
  for (int it = 0; it < NIT; ++it) {
    const int buf = it & 1;
    const bool more = (it + 1) < NIT;
    if (more) issueA((it + 1) * GBK, buf ^ 1);  // prefetch next A tile (async)
    storeB();                                   // stage current B tile into LDS
    if (more) loadB((it + 1) * GBK);            // prefetch next B tile into regs
    if (more) { S_WAIT_ASYNC("0x2"); } else { S_WAIT_ASYNC("0x0"); }
    __syncthreads();

    v16bf af[4], bfr[2];
    #pragma unroll
    for (int i = 0; i < 4; ++i) af[i] = load_frag(&As[buf][wm + 16 * i + nlane][kb]);
    #pragma unroll
    for (int j = 0; j < 2; ++j) bfr[j] = load_frag(&Bs[wn + 16 * j + nlane][kb]);
    #pragma unroll
    for (int i = 0; i < 4; ++i)
      #pragma unroll
      for (int j = 0; j < 2; ++j)
        acc[i][j] = wmma_bf16(af[i], bfr[j], acc[i][j]);
    __syncthreads();
  }

  // ---- epilogue: C layout = (m = r + moff, n = nlane) per 16x16 tile
  #pragma unroll
  for (int i = 0; i < 4; ++i) {
    #pragma unroll
    for (int j = 0; j < 2; ++j) {
      int n = n0 + wn + 16 * j + nlane;
      float bv = bias[n];
      #pragma unroll
      for (int r = 0; r < 8; ++r) {
        int m = m0 + wm + 16 * i + r + moff;
        float val = acc[i][j][r] + bv;
        if (OUT_F32) outf[(size_t)m * ldc + n] = val;
        else         outb[(size_t)m * ldc + n] = (__bf16)val;
      }
    }
  }
}

// ---------------------------------------------------------------------------
// Kernel 3: pos[h][q][k] = bias[q][k][:] @ W_bias[:,h] + b_bias[h]
// ---------------------------------------------------------------------------
__global__ __launch_bounds__(256) void pos_bias_kernel(
    const float* __restrict__ bias, const float* __restrict__ Wb,
    const float* __restrict__ bb, float* __restrict__ pos) {
  __shared__ float w[16][16];
  __shared__ float bbs[16];
  int tid = threadIdx.x;
  w[tid >> 4][tid & 15] = Wb[tid];          // Wb[d][h], 256 elems
  if (tid < 16) bbs[tid] = bb[tid];
  __syncthreads();

  int q = blockIdx.y;
  int k = blockIdx.x * 256 + tid;
  const float* bp = bias + ((size_t)q * 1024 + k) * 16;
  float bv[16];
  #pragma unroll
  for (int d = 0; d < 16; ++d) bv[d] = bp[d];
  #pragma unroll
  for (int h = 0; h < 16; ++h) {
    float a = bbs[h];
    #pragma unroll
    for (int d = 0; d < 16; ++d) a += bv[d] * w[d][h];
    pos[((size_t)h << 20) + (size_t)q * 1024 + k] = a;
  }
}

// ---------------------------------------------------------------------------
// Kernel 4: flash attention. grid (B*H, T/64), 128 threads = 4 waves.
// Q: async prologue. K: async double-buffered. V: register-staged transpose.
// ---------------------------------------------------------------------------
#define FPAD 72   // 64 + 8 (144B row stride -> conflict-free, 16B aligned)

__global__ __launch_bounds__(128) void flash_attn(
    const __bf16* __restrict__ qkv, const float* __restrict__ pos,
    __bf16* __restrict__ y) {
  __shared__ __align__(16) __bf16 Qs[64][FPAD];
  __shared__ __align__(16) __bf16 Ks[2][64][FPAD];
  __shared__ __align__(16) __bf16 Vt[64][FPAD];       // [d][j] (transposed)
  __shared__ __align__(16) __bf16 Ps[4][16][FPAD];    // per-wave P staging

  const int bh = blockIdx.x;
  const int b = bh >> 4, h = bh & 15;
  const int q0 = blockIdx.y * 64;
  const __bf16* qb  = qkv + (size_t)b * 1024 * 3072 + h * 64;
  const __bf16* kbp = qb + 1024;
  const __bf16* vbp = qb + 2048;
  const float* prow = pos + ((size_t)h << 20);

  const int tid  = threadIdx.x;
  const int wave = tid >> 5;
  const int lane = tid & 31;
  const int kb = (lane < 16) ? 0 : 8;
  const int nlane = lane & 15;
  const int moff = (lane >= 16) ? 8 : 0;

  // 64x64 tile = 512 x 16B chunks, 4 per thread
  const int tr = tid >> 3, tc = (tid & 7) * 8;
  auto issueK = [&](int kt, int bufi) {
    #pragma unroll
    for (int i = 0; i < 4; ++i) {
      int r = tr + 16 * i;
      async_copy_b128(&kbp[(size_t)(kt + r) * 3072 + tc], &Ks[bufi][r][tc]);
    }
  };
  // V: thread owns global row j = tid>>1, 32 cols
  const int vj = tid >> 1, vc0 = (tid & 1) * 32;
  U8 vr[4];
  auto loadV = [&](int kt) {
    #pragma unroll
    for (int cc = 0; cc < 4; ++cc)
      vr[cc].u = *(const uint4*)&vbp[(size_t)(kt + vj) * 3072 + vc0 + cc * 8];
  };
  auto storeV = [&]() {
    #pragma unroll
    for (int cc = 0; cc < 4; ++cc)
      #pragma unroll
      for (int e = 0; e < 8; ++e) Vt[vc0 + cc * 8 + e][vj] = vr[cc].h[e];
  };

  // ---- Q prologue (async)
  #pragma unroll
  for (int i = 0; i < 4; ++i) {
    int r = tr + 16 * i;
    async_copy_b128(&qb[(size_t)(q0 + r) * 3072 + tc], &Qs[r][tc]);
  }
  S_WAIT_ASYNC("0x0");
  __syncthreads();

  const int mrow = wave * 16 + nlane;
  v16bf qf0 = load_frag(&Qs[mrow][kb]);        // d = 0..31
  v16bf qf1 = load_frag(&Qs[mrow][kb + 32]);   // d = 32..63

  float mstat[8], lstat[8];
  #pragma unroll
  for (int r = 0; r < 8; ++r) { mstat[r] = -1e30f; lstat[r] = 0.f; }
  v8f o[4] = {};

  issueK(0, 0);
  loadV(0);

  for (int ti = 0; ti < 16; ++ti) {
    const int kt = ti * 64;
    const int buf = ti & 1;
    const bool more = ti < 15;
    if (more) issueK(kt + 64, buf ^ 1);  // prefetch next K tile (async)
    storeV();                            // stage current V tile (transposed)
    if (more) loadV(kt + 64);            // prefetch next V tile into regs
    if (more) { S_WAIT_ASYNC("0x4"); } else { S_WAIT_ASYNC("0x0"); }
    __syncthreads();

    // ---- S = (q k^T) * 1/sqrt(64) + pos_bias   (4 tiles of 16x16)
    v8f s[4];
    #pragma unroll
    for (int jn = 0; jn < 4; ++jn) {
      v16bf kf0 = load_frag(&Ks[buf][jn * 16 + nlane][kb]);
      v16bf kf1 = load_frag(&Ks[buf][jn * 16 + nlane][kb + 32]);
      v8f c = {};
      c = wmma_bf16(qf0, kf0, c);
      c = wmma_bf16(qf1, kf1, c);
      s[jn] = c;
    }
    #pragma unroll
    for (int jn = 0; jn < 4; ++jn) {
      #pragma unroll
      for (int r = 0; r < 8; ++r) {
        int qq = q0 + wave * 16 + r + moff;
        s[jn][r] = s[jn][r] * 0.125f +
                   prow[(size_t)qq * 1024 + kt + jn * 16 + nlane];
      }
    }

    // ---- online softmax (row = half-wave; butterfly over 16 lanes)
    float rmax[8];
    #pragma unroll
    for (int r = 0; r < 8; ++r)
      rmax[r] = fmaxf(fmaxf(s[0][r], s[1][r]), fmaxf(s[2][r], s[3][r]));
    #pragma unroll
    for (int d = 1; d < 16; d <<= 1)
      #pragma unroll
      for (int r = 0; r < 8; ++r)
        rmax[r] = fmaxf(rmax[r], __shfl_xor(rmax[r], d, 32));

    float scalef[8], rsum[8];
    #pragma unroll
    for (int r = 0; r < 8; ++r) {
      float mn = fmaxf(mstat[r], rmax[r]);
      scalef[r] = __expf(mstat[r] - mn);
      mstat[r] = mn;
      rsum[r] = 0.f;
    }
    #pragma unroll
    for (int jn = 0; jn < 4; ++jn)
      #pragma unroll
      for (int r = 0; r < 8; ++r) {
        float p = __expf(s[jn][r] - mstat[r]);
        s[jn][r] = p;
        rsum[r] += p;
      }
    #pragma unroll
    for (int d = 1; d < 16; d <<= 1)
      #pragma unroll
      for (int r = 0; r < 8; ++r) rsum[r] += __shfl_xor(rsum[r], d, 32);
    #pragma unroll
    for (int r = 0; r < 8; ++r) {
      lstat[r] = lstat[r] * scalef[r] + rsum[r];
      #pragma unroll
      for (int dn = 0; dn < 4; ++dn) o[dn][r] *= scalef[r];
    }

    // ---- stage P (bf16) through per-wave LDS: C-layout -> A-layout
    #pragma unroll
    for (int jn = 0; jn < 4; ++jn)
      #pragma unroll
      for (int r = 0; r < 8; ++r)
        Ps[wave][r + moff][jn * 16 + nlane] = (__bf16)s[jn][r];
    // same-wave LDS ops are in-order; no barrier needed
    v16bf pf0 = load_frag(&Ps[wave][nlane][kb]);        // j = 0..31
    v16bf pf1 = load_frag(&Ps[wave][nlane][kb + 32]);   // j = 32..63

    // ---- O += P @ V
    #pragma unroll
    for (int dn = 0; dn < 4; ++dn) {
      v16bf vf0 = load_frag(&Vt[dn * 16 + nlane][kb]);
      v16bf vf1 = load_frag(&Vt[dn * 16 + nlane][kb + 32]);
      o[dn] = wmma_bf16(pf0, vf0, o[dn]);
      o[dn] = wmma_bf16(pf1, vf1, o[dn]);
    }
    __syncthreads();
  }

  // ---- finalize: O /= l, write y[b][t][h*64+d] (bf16)
  #pragma unroll
  for (int r = 0; r < 8; ++r) {
    float inv = 1.0f / lstat[r];
    int t = q0 + wave * 16 + r + moff;
    size_t base = ((size_t)(b * 1024 + t)) * 1024 + h * 64;
    #pragma unroll
    for (int dn = 0; dn < 4; ++dn)
      y[base + dn * 16 + nlane] = (__bf16)(o[dn][r] * inv);
  }
}

// ---------------------------------------------------------------------------
// Launch
// ---------------------------------------------------------------------------
extern "C" void kernel_launch(void* const* d_in, const int* in_sizes, int n_in,
                              void* d_out, int out_size, void* d_ws, size_t ws_size,
                              hipStream_t stream) {
  const float* x      = (const float*)d_in[0];
  const float* bias   = (const float*)d_in[1];
  const float* W_attn = (const float*)d_in[2];
  const float* b_attn = (const float*)d_in[3];
  const float* W_proj = (const float*)d_in[4];
  const float* b_proj = (const float*)d_in[5];
  const float* W_bias = (const float*)d_in[6];
  const float* b_bias = (const float*)d_in[7];
  float* out = (float*)d_out;

  char* ws = (char*)d_ws;
  __bf16* xb  = (__bf16*)(ws);                          // 16 MiB
  __bf16* wab = (__bf16*)(ws + (size_t)16  * (1u<<20)); //  6 MiB
  __bf16* wpb = (__bf16*)(ws + (size_t)22  * (1u<<20)); //  2 MiB
  __bf16* qkv = (__bf16*)(ws + (size_t)24  * (1u<<20)); // 48 MiB
  float*  pos = (float*) (ws + (size_t)72  * (1u<<20)); // 64 MiB
  __bf16* yb  = (__bf16*)(ws + (size_t)136 * (1u<<20)); // 16 MiB

  // 1) fp32 -> bf16 conversions
  f32_to_bf16<<<(8192 * 1024) / 1024, 256, 0, stream>>>(x, xb, 8192 * 1024);
  f32_to_bf16<<<(1024 * 3072) / 1024, 256, 0, stream>>>(W_attn, wab, 1024 * 3072);
  f32_to_bf16<<<(1024 * 1024) / 1024, 256, 0, stream>>>(W_proj, wpb, 1024 * 1024);

  // 2) qkv = x @ W_attn + b_attn   (M=8192, N=3072, K=1024) -> bf16
  gemm_bf16<false><<<dim3(3072 / GBN, 8192 / GBM), 256, 0, stream>>>(
      xb, 1024, wab, 3072, b_attn, qkv, nullptr, 3072, 1024);

  // 3) pos[h][q][k] = bias @ W_bias + b_bias
  pos_bias_kernel<<<dim3(1024 / 256, 1024), 256, 0, stream>>>(bias, W_bias, b_bias, pos);

  // 4) flash attention -> y (bf16, [B*T][1024], head-interleaved)
  flash_attn<<<dim3(8 * 16, 1024 / 64), 128, 0, stream>>>(qkv, pos, yb);

  // 5) out = y @ W_proj + b_proj   (M=8192, N=1024, K=1024) -> fp32
  gemm_bf16<true><<<dim3(1024 / GBN, 8192 / GBM), 256, 0, stream>>>(
      yb, 1024, wpb, 1024, b_proj, nullptr, out, 1024, 1024);
}
